// SUTBlock_35562329211359
// MI455X (gfx1250) — matrix operand order repaired and checked by reading, not compiled, not run on hardware
//
#include <hip/hip_runtime.h>
#include <hip/hip_bf16.h>

#define USE_TDM 1

typedef __bf16 bf16_t;
typedef __attribute__((ext_vector_type(8)))  __bf16 bf8v;
typedef __attribute__((ext_vector_type(16))) __bf16 bf16v;
typedef __attribute__((ext_vector_type(8)))  float  f8v;
typedef __attribute__((ext_vector_type(4)))  unsigned int v4u;
typedef __attribute__((ext_vector_type(8)))  int v8i;
typedef __attribute__((ext_vector_type(4)))  int v4i;

#define T_TOK 4096
#define H_DIM 1024
#define NEG_BIG (-3e38f)

__device__ __forceinline__ bf16v mk_frag(const bf16_t* lo, const bf16_t* hi) {
  bf8v a = *(const bf8v*)lo;
  bf8v b = *(const bf8v*)hi;
  return __builtin_shufflevector(a, b, 0,1,2,3,4,5,6,7,8,9,10,11,12,13,14,15);
}
__device__ __forceinline__ f8v wmma_bf16(bf16v a, bf16v b, f8v c) {
  // D = A(16x32 bf16) x B(32x16 bf16) + C(16x16 f32)
  return __builtin_amdgcn_wmma_f32_16x16x32_bf16(false, a, false, b, (short)0, c, false, false);
}
__device__ __forceinline__ f8v zero8() {
  f8v z;
#pragma unroll
  for (int i = 0; i < 8; ++i) z[i] = 0.f;
  return z;
}

#if USE_TDM
// ---------------------------------------------------------------------------
// TDM: DMA a 2D tile [128 rows x 32 bf16] (row-major, row stride `stride0`
// elems) from global into LDS with hardware padding of 16B per 64B row
// -> LDS row stride 80B (= 40 bf16), matching the fragment-read layout.
// D# layout per CDNA5 ISA ch.8 (group0/group1; groups 2-4 zero for 2D tile).
// 6-arg builtin form (clang-23 / amdgpu-toolchain).
// ---------------------------------------------------------------------------
__device__ __forceinline__ void tdm_load_tile_128x32(
    const bf16_t* tile_base, unsigned lds_byte_addr,
    unsigned tensor_d0, unsigned long long stride0_elems) {
  unsigned long long ga = (unsigned long long)(size_t)tile_base;
  v4u g0;
  g0[0] = 1u;                                   // count=1, user mode
  g0[1] = lds_byte_addr;                        // lds_addr (bytes)
  g0[2] = (unsigned)ga;                         // global_addr[31:0]
  g0[3] = ((unsigned)(ga >> 32) & 0x01FFFFFFu)  // global_addr[56:32]
          | (2u << 30);                         // type = 2 ("image")
  v8i g1;
  g1[0] = (int)((1u << 16)      // data_size = 1 -> 2 bytes
              | (1u << 20)      // pad_enable
              | (3u << 22)      // pad_interval: 16 DWORDs = 64B
              | (3u << 25));    // pad_amount: 4 DWORDs = 16B
  g1[1] = (int)((tensor_d0 & 0xFFFFu) << 16);                       // dim0 lo16
  g1[2] = (int)((tensor_d0 >> 16) | (128u << 16));                  // dim0 hi | dim1(=128) lo16
  g1[3] = (int)(0u | (32u << 16));                                  // dim1 hi | tile_dim0=32
  g1[4] = (int)(128u);                                              // tile_dim1=128, tile_dim2=0
  g1[5] = (int)(unsigned)(stride0_elems & 0xFFFFFFFFu);             // stride0[31:0]
  g1[6] = (int)(unsigned)(stride0_elems >> 32);                     // stride0[47:32]
  g1[7] = 0;
  v4i g2; g2[0] = 0; g2[1] = 0; g2[2] = 0; g2[3] = 0;
  v4i g3; g3[0] = 0; g3[1] = 0; g3[2] = 0; g3[3] = 0;
  v8i g4;
#pragma unroll
  for (int i = 0; i < 8; ++i) g4[i] = 0;
  __builtin_amdgcn_tensor_load_to_lds(g0, g1, g2, g3, g4, 0);
}
#endif

// ---------------------------------------------------------------------------
// Weight transpose + fp32 -> bf16 convert:  src [R,C] f32 -> dst [C,R] bf16
// ---------------------------------------------------------------------------
__global__ __launch_bounds__(256) void transpose_f32_to_bf16(
    const float* __restrict__ src, bf16_t* __restrict__ dst, int R, int C) {
  __shared__ float tile[32][33];
  size_t off = (size_t)blockIdx.z * R * C;
  int c0 = blockIdx.x * 32, r0 = blockIdx.y * 32;
  for (int j = threadIdx.y; j < 32; j += 8)
    tile[j][threadIdx.x] = src[off + (size_t)(r0 + j) * C + c0 + threadIdx.x];
  __syncthreads();
  for (int j = threadIdx.y; j < 32; j += 8)
    dst[off + (size_t)(c0 + j) * R + r0 + threadIdx.x] = (bf16_t)tile[threadIdx.x][j];
}

// ---------------------------------------------------------------------------
// RMSNorm: x f32 [T,1024] -> out bf16; optionally copy x (pre-norm) to copy_out
// ---------------------------------------------------------------------------
__global__ __launch_bounds__(256) void rmsnorm_kernel(
    const float* __restrict__ x, const float* __restrict__ w,
    bf16_t* __restrict__ out, float* __restrict__ copy_out) {
  __shared__ float red[256];
  int tid = threadIdx.x;
  size_t base = (size_t)blockIdx.x * H_DIM;
  float v0 = x[base + tid], v1 = x[base + tid + 256];
  float v2 = x[base + tid + 512], v3 = x[base + tid + 768];
  red[tid] = v0 * v0 + v1 * v1 + v2 * v2 + v3 * v3;
  __syncthreads();
  for (int s = 128; s > 0; s >>= 1) {
    if (tid < s) red[tid] += red[tid + s];
    __syncthreads();
  }
  float rstd = rsqrtf(red[0] * (1.f / 1024.f) + 1e-5f);
  out[base + tid]       = (bf16_t)(v0 * rstd * w[tid]);
  out[base + tid + 256] = (bf16_t)(v1 * rstd * w[tid + 256]);
  out[base + tid + 512] = (bf16_t)(v2 * rstd * w[tid + 512]);
  out[base + tid + 768] = (bf16_t)(v3 * rstd * w[tid + 768]);
  if (copy_out) {
    copy_out[base + tid] = v0; copy_out[base + tid + 256] = v1;
    copy_out[base + tid + 512] = v2; copy_out[base + tid + 768] = v3;
  }
}

// ---------------------------------------------------------------------------
// Tiled WMMA GEMM: C[M,N] = A[M,K] (bf16, lda) x BT[N,K] (bf16, ldb=K)
// Block 256 thr (8 waves), tile 128x128, wave = 64x32 (4x2 16x16 frags).
// TDM path: double-buffered LDS; wave 0 DMAs tile k+1 (tensor_load_to_lds,
// TENSORcnt) while all waves compute tile k. 1 barrier / k-step.
// epi: 0 store bf16; 1 gelu->bf16; 2 f32 = resid + c; 3 f32 += gate[row,gcol]*c
// ---------------------------------------------------------------------------
__global__ __launch_bounds__(256) void gemm_bf16_wmma(
    const bf16_t* __restrict__ A, int lda, const bf16_t* __restrict__ BT,
    int N, int K, int epi,
    bf16_t* __restrict__ Cb, float* __restrict__ Cf,
    const float* __restrict__ resid, const float* __restrict__ gate, int gcol) {
#if USE_TDM
  __shared__ __align__(16) bf16_t lA[2][128 * 40];
  __shared__ __align__(16) bf16_t lB[2][128 * 40];
#else
  __shared__ __align__(16) bf16_t lA[1][128 * 40];
  __shared__ __align__(16) bf16_t lB[1][128 * 40];
#endif
  const int tid = threadIdx.x;
  const int lane = tid & 31;
  const int lh = lane & 15, hi = lane >> 4;
  const int wid = tid >> 5;
  const int wm = (wid & 1) * 64, wn = (wid >> 1) * 32;
  const int m0 = blockIdx.y * 128, n0 = blockIdx.x * 128;

  f8v acc[4][2];
#pragma unroll
  for (int i = 0; i < 4; ++i)
#pragma unroll
    for (int j = 0; j < 2; ++j) acc[i][j] = zero8();

#if USE_TDM
  const int nk = K >> 5;
  const bf16_t* Abase = A + (size_t)m0 * lda;
  const bf16_t* Bbase = BT + (size_t)n0 * K;
  if (wid == 0) {
    tdm_load_tile_128x32(Abase, (unsigned)(size_t)&lA[0][0], (unsigned)K,
                         (unsigned long long)lda);
    tdm_load_tile_128x32(Bbase, (unsigned)(size_t)&lB[0][0], (unsigned)K,
                         (unsigned long long)K);
    __builtin_amdgcn_s_wait_tensorcnt(0);
  }
  __syncthreads();
  for (int kt = 0; kt < nk; ++kt) {
    const int cur = kt & 1;
    if (wid == 0 && kt + 1 < nk) {
      const int k2 = (kt + 1) << 5;
      tdm_load_tile_128x32(Abase + k2, (unsigned)(size_t)&lA[1 - cur][0],
                           (unsigned)K, (unsigned long long)lda);
      tdm_load_tile_128x32(Bbase + k2, (unsigned)(size_t)&lB[1 - cur][0],
                           (unsigned)K, (unsigned long long)K);
    }
    bf16v af[4], bfrag[2];
#pragma unroll
    for (int mi = 0; mi < 4; ++mi) {
      const bf16_t* p = &lA[cur][(wm + mi * 16 + lh) * 40 + hi * 8];
      af[mi] = mk_frag(p, p + 16);
    }
#pragma unroll
    for (int ni = 0; ni < 2; ++ni) {
      const bf16_t* p = &lB[cur][(wn + ni * 16 + lh) * 40 + hi * 8];
      bfrag[ni] = mk_frag(p, p + 16);
    }
#pragma unroll
    for (int mi = 0; mi < 4; ++mi)
#pragma unroll
      for (int ni = 0; ni < 2; ++ni)
        acc[mi][ni] = wmma_bf16(af[mi], bfrag[ni], acc[mi][ni]);
    if (wid == 0) __builtin_amdgcn_s_wait_tensorcnt(0);
    __syncthreads();
  }
#else
  const int lr = tid >> 1, lc = (tid & 1) * 16;
  for (int k0 = 0; k0 < K; k0 += 32) {
    const bf16_t* ga = A + (size_t)(m0 + lr) * lda + k0 + lc;
    bf8v a0 = *(const bf8v*)ga, a1 = *(const bf8v*)(ga + 8);
    const bf16_t* gb = BT + (size_t)(n0 + lr) * K + k0 + lc;
    bf8v b0 = *(const bf8v*)gb, b1 = *(const bf8v*)(gb + 8);
    *(bf8v*)&lA[0][lr * 40 + lc] = a0; *(bf8v*)&lA[0][lr * 40 + lc + 8] = a1;
    *(bf8v*)&lB[0][lr * 40 + lc] = b0; *(bf8v*)&lB[0][lr * 40 + lc + 8] = b1;
    __syncthreads();
    bf16v af[4], bfrag[2];
#pragma unroll
    for (int mi = 0; mi < 4; ++mi) {
      const bf16_t* p = &lA[0][(wm + mi * 16 + lh) * 40 + hi * 8];
      af[mi] = mk_frag(p, p + 16);
    }
#pragma unroll
    for (int ni = 0; ni < 2; ++ni) {
      const bf16_t* p = &lB[0][(wn + ni * 16 + lh) * 40 + hi * 16];
      bfrag[ni] = mk_frag(p, p + 8);
    }
#pragma unroll
    for (int mi = 0; mi < 4; ++mi)
#pragma unroll
      for (int ni = 0; ni < 2; ++ni)
        acc[mi][ni] = wmma_bf16(af[mi], bfrag[ni], acc[mi][ni]);
    __syncthreads();
  }
#endif

#pragma unroll
  for (int mi = 0; mi < 4; ++mi)
#pragma unroll
    for (int ni = 0; ni < 2; ++ni)
#pragma unroll
      for (int v = 0; v < 8; ++v) {
        int row = m0 + wm + mi * 16 + v + 8 * hi;
        int col = n0 + wn + ni * 16 + lh;
        float val = acc[mi][ni][v];
        size_t idx = (size_t)row * N + col;
        if (epi == 0) {
          Cb[idx] = (bf16_t)val;
        } else if (epi == 1) {
          Cb[idx] = (bf16_t)(0.5f * val * (1.f + erff(val * 0.70710678118654752f)));
        } else if (epi == 2) {
          Cf[idx] = resid[idx] + val;
        } else {
          Cf[idx] += gate[(size_t)row * 8 + gcol] * val;
        }
      }
}

// ---------------------------------------------------------------------------
// Causal flash attention. qkv bf16 [T, 3*1024]; out bf16 [T,1024].
// grid (S/64, B*NH), 128 thr. Wave owns 16 queries; computes S^T = K x Q^T so
// each lane owns one query; S^T D-frag layout == A-frag layout of P for P x V.
// ---------------------------------------------------------------------------
__global__ __launch_bounds__(128) void attn_kernel(
    const bf16_t* __restrict__ qkv, bf16_t* __restrict__ ao) {
  __shared__ __align__(16) bf16_t lK[32 * 72];
  __shared__ __align__(16) bf16_t lVT[64 * 40];
  const int b = blockIdx.y >> 4, h = blockIdx.y & 15;
  const int q0blk = blockIdx.x * 64;
  const int lane = threadIdx.x & 31, wid = threadIdx.x >> 5;
  const int lh = lane & 15, hi = lane >> 4;
  const int q0 = q0blk + wid * 16;
  const size_t rs = 3072;
  const size_t brow = (size_t)b * 2048;

  bf16v qf[2];  // B-frag of Q^T: lane = query lh, elems = hd 32c+16hi..+15
  {
    const bf16_t* qp = qkv + (brow + q0 + lh) * rs + h * 64;
    qf[0] = mk_frag(qp + hi * 16, qp + hi * 16 + 8);
    qf[1] = mk_frag(qp + 32 + hi * 16, qp + 32 + hi * 16 + 8);
  }

  f8v o[4];
#pragma unroll
  for (int i = 0; i < 4; ++i) o[i] = zero8();
  float m_run = NEG_BIG, l_run = 0.f;

  const int nkb = (q0blk + 64) >> 5;
  for (int kb = 0; kb < nkb; ++kb) {
    const int k0 = kb << 5;
    __syncthreads();
    {  // stage K [32x64] and V^T [64x32] for this head
      int r = threadIdx.x >> 2, cb = (threadIdx.x & 3) << 4;
      const bf16_t* kp = qkv + (brow + k0 + r) * rs + 1024 + h * 64 + cb;
      bf8v ka = *(const bf8v*)kp, kb8 = *(const bf8v*)(kp + 8);
      *(bf8v*)&lK[r * 72 + cb] = ka;
      *(bf8v*)&lK[r * 72 + cb + 8] = kb8;
      const bf16_t* vp = qkv + (brow + k0 + r) * rs + 2048 + h * 64 + cb;
      bf8v va = *(const bf8v*)vp, vb = *(const bf8v*)(vp + 8);
#pragma unroll
      for (int i = 0; i < 8; ++i) {
        lVT[(cb + i) * 40 + r] = va[i];
        lVT[(cb + 8 + i) * 40 + r] = vb[i];
      }
    }
    __syncthreads();
    if (q0 + 15 >= k0) {  // wave-uniform
      f8v sT[2];
#pragma unroll
      for (int blk = 0; blk < 2; ++blk) {
        f8v s = zero8();
#pragma unroll
        for (int c = 0; c < 2; ++c) {
          const bf16_t* p = &lK[(blk * 16 + lh) * 72 + c * 32 + hi * 16];
          bf16v a = mk_frag(p, p + 8);  // A-frag of K tile (keys x hd)
          s = wmma_bf16(a, qf[c], s);
        }
        sT[blk] = s;
      }
      // mask + online softmax; lane owns query q0+lh (both halves agree)
      float p[2][8];
      float mloc = NEG_BIG;
      int qg = q0 + lh;
#pragma unroll
      for (int blk = 0; blk < 2; ++blk)
#pragma unroll
        for (int v = 0; v < 8; ++v) {
          int key = k0 + blk * 16 + v + 8 * hi;
          float sv = sT[blk][v] * 0.125f;
          if (key > qg) sv = NEG_BIG;
          p[blk][v] = sv;
          mloc = fmaxf(mloc, sv);
        }
      mloc = fmaxf(mloc, __shfl_xor(mloc, 16));
      float m_new = fmaxf(m_run, mloc);
      float alpha = __expf(m_run - m_new);
      float rsum = 0.f;
#pragma unroll
      for (int blk = 0; blk < 2; ++blk)
#pragma unroll
        for (int v = 0; v < 8; ++v) {
          float ev = __expf(p[blk][v] - m_new);
          p[blk][v] = ev;
          rsum += ev;
        }
      rsum += __shfl_xor(rsum, 16);
      l_run = l_run * alpha + rsum;
      m_run = m_new;
      // P in S^T D-frag layout == A-frag of P (16 queries x 32 keys). Free!
      bf16v pa;
#pragma unroll
      for (int v = 0; v < 8; ++v) {
        pa[v] = (bf16_t)p[0][v];
        pa[v + 8] = (bf16_t)p[1][v];
      }
      float f[8];  // O-frag row v+8hi is query v+8hi -> fetch its alpha
#pragma unroll
      for (int v = 0; v < 8; ++v) f[v] = __shfl(alpha, v + 8 * hi);
#pragma unroll
      for (int ni = 0; ni < 4; ++ni) {
        const bf16_t* vp = &lVT[(ni * 16 + lh) * 40 + hi * 8];
        bf16v bv = mk_frag(vp, vp + 16);  // B-frag of V (keys x hd)
        f8v c;
#pragma unroll
        for (int v = 0; v < 8; ++v) c[v] = o[ni][v] * f[v];
        o[ni] = wmma_bf16(pa, bv, c);
      }
    }
  }
  float linv = 1.f / l_run;
  float fn[8];
#pragma unroll
  for (int v = 0; v < 8; ++v) fn[v] = __shfl(linv, v + 8 * hi);
#pragma unroll
  for (int ni = 0; ni < 4; ++ni)
#pragma unroll
    for (int v = 0; v < 8; ++v)
      ao[(brow + q0 + v + 8 * hi) * (size_t)H_DIM + h * 64 + ni * 16 + lh] =
          (bf16_t)(o[ni][v] * fn[v]);
}

// ---------------------------------------------------------------------------
// Router: logits = y @ w_router [T,8]; softmax; top-2; gate; freq (atomics of
// exact 1.0f -> deterministic). 32 tokens per block, thread = (token, expert).
// ---------------------------------------------------------------------------
__global__ __launch_bounds__(256) void router_kernel(
    const bf16_t* __restrict__ y, const float* __restrict__ wr,
    float* __restrict__ logits, float* __restrict__ gate, float* __restrict__ freq) {
  __shared__ float lg[32][8];
  int e = threadIdx.x & 7, tl = threadIdx.x >> 3;
  int tok = blockIdx.x * 32 + tl;
  const bf16_t* yr = y + (size_t)tok * H_DIM;
  float acc = 0.f;
  for (int hh = 0; hh < H_DIM; hh += 4) {
    acc += (float)yr[hh] * wr[hh * 8 + e];
    acc += (float)yr[hh + 1] * wr[(hh + 1) * 8 + e];
    acc += (float)yr[hh + 2] * wr[(hh + 2) * 8 + e];
    acc += (float)yr[hh + 3] * wr[(hh + 3) * 8 + e];
  }
  lg[tl][e] = acc;
  logits[(size_t)tok * 8 + e] = acc;
  __syncthreads();
  if (threadIdx.x < 32) {
    int t2 = blockIdx.x * 32 + (int)threadIdx.x;
    float l[8];
#pragma unroll
    for (int i = 0; i < 8; ++i) l[i] = lg[threadIdx.x][i];
    int i1 = 0; float b1 = l[0];
#pragma unroll
    for (int i = 1; i < 8; ++i) if (l[i] > b1) { b1 = l[i]; i1 = i; }
    int i2 = -1; float b2 = NEG_BIG;
#pragma unroll
    for (int i = 0; i < 8; ++i) if (i != i1 && l[i] > b2) { b2 = l[i]; i2 = i; }
    float e2 = __expf(b2 - b1);
    float g1 = 1.f / (1.f + e2), g2 = e2 / (1.f + e2);
#pragma unroll
    for (int i = 0; i < 8; ++i)
      gate[(size_t)t2 * 8 + i] = (i == i1) ? g1 : ((i == i2) ? g2 : 0.f);
    atomicAdd(&freq[i1], 1.f);
    atomicAdd(&freq[i2], 1.f);
  }
}

__global__ void zero_freq_kernel(float* freq) {
  if (threadIdx.x < 8) freq[threadIdx.x] = 0.f;
}

// ---------------------------------------------------------------------------
extern "C" void kernel_launch(void* const* d_in, const int* in_sizes, int n_in,
                              void* d_out, int out_size, void* d_ws, size_t ws_size,
                              hipStream_t stream) {
  (void)in_sizes; (void)n_in; (void)out_size; (void)ws_size;
  const float* x    = (const float*)d_in[0];
  const float* ln1  = (const float*)d_in[1];
  const float* ln2  = (const float*)d_in[2];
  const float* wqkv = (const float*)d_in[3];
  const float* wo   = (const float*)d_in[4];
  const float* wrt  = (const float*)d_in[5];
  const float* w1   = (const float*)d_in[6];
  const float* w2   = (const float*)d_in[7];
  float* out = (float*)d_out;

  char* ws = (char*)d_ws;
  size_t off = 0;
  auto take = [&](size_t bytes) {
    char* p = ws + off;
    off += (bytes + 255) & ~(size_t)255;
    return p;
  };
  bf16_t* wqkvT = (bf16_t*)take((size_t)3072 * 1024 * 2);
  bf16_t* woT   = (bf16_t*)take((size_t)1024 * 1024 * 2);
  bf16_t* w1T   = (bf16_t*)take((size_t)8 * 2048 * 1024 * 2);
  bf16_t* w2T   = (bf16_t*)take((size_t)8 * 1024 * 2048 * 2);
  bf16_t* xn    = (bf16_t*)take((size_t)T_TOK * 1024 * 2);
  bf16_t* qkvb  = (bf16_t*)take((size_t)T_TOK * 3072 * 2);
  bf16_t* attnb = (bf16_t*)take((size_t)T_TOK * 1024 * 2);
  float*  hbuf  = (float*)take((size_t)T_TOK * 1024 * 4);
  bf16_t* yb    = (bf16_t*)take((size_t)T_TOK * 1024 * 2);
  float*  gate  = (float*)take((size_t)T_TOK * 8 * 4);
  bf16_t* h1b   = (bf16_t*)take((size_t)T_TOK * 2048 * 2);

  float* logits = out + (size_t)T_TOK * 1024;
  float* freq   = logits + (size_t)T_TOK * 8;

  dim3 tb(32, 8);
  transpose_f32_to_bf16<<<dim3(96, 32, 1), tb, 0, stream>>>(wqkv, wqkvT, 1024, 3072);
  transpose_f32_to_bf16<<<dim3(32, 32, 1), tb, 0, stream>>>(wo, woT, 1024, 1024);
  transpose_f32_to_bf16<<<dim3(64, 32, 8), tb, 0, stream>>>(w1, w1T, 1024, 2048);
  transpose_f32_to_bf16<<<dim3(32, 64, 8), tb, 0, stream>>>(w2, w2T, 2048, 1024);

  // h = x + attn(rmsnorm(x)) ; qkv/attn in bf16 WMMA
  rmsnorm_kernel<<<T_TOK, 256, 0, stream>>>(x, ln1, xn, nullptr);
  gemm_bf16_wmma<<<dim3(24, 32), 256, 0, stream>>>(xn, 1024, wqkvT, 3072, 1024, 0,
                                                   qkvb, nullptr, nullptr, nullptr, 0);
  attn_kernel<<<dim3(32, 32), 128, 0, stream>>>(qkvb, attnb);
  gemm_bf16_wmma<<<dim3(8, 32), 256, 0, stream>>>(attnb, 1024, woT, 1024, 1024, 2,
                                                  nullptr, hbuf, x, nullptr, 0);
  // y = rmsnorm(h); also seed d_out[0:T*H] = h for MoE accumulation
  rmsnorm_kernel<<<T_TOK, 256, 0, stream>>>(hbuf, ln2, yb, out);
  zero_freq_kernel<<<1, 32, 0, stream>>>(freq);
  router_kernel<<<T_TOK / 32, 256, 0, stream>>>(yb, wrt, logits, gate, freq);
  // MoE: out += gate[:,e] * (gelu(y @ w1[e]) @ w2[e]), sequential per expert
  for (int e = 0; e < 8; ++e) {
    const bf16_t* w1e = w1T + (size_t)e * 2048 * 1024;
    const bf16_t* w2e = w2T + (size_t)e * 1024 * 2048;
    gemm_bf16_wmma<<<dim3(16, 32), 256, 0, stream>>>(yb, 1024, w1e, 2048, 1024, 1,
                                                     h1b, nullptr, nullptr, nullptr, 0);
    gemm_bf16_wmma<<<dim3(8, 32), 256, 0, stream>>>(h1b, 2048, w2e, 1024, 2048, 3,
                                                    nullptr, out, nullptr, gate, e);
  }
}
